// MDCSpikeActor_21938692948165
// MI455X (gfx1250) — matrix operand order repaired and checked by reading, not compiled, not run on hardware
//
#include <hip/hip_runtime.h>
#include <hip/hip_bf16.h>

typedef __attribute__((ext_vector_type(16))) __bf16 v16bf;
typedef __attribute__((ext_vector_type(8)))  float  v8f;

#define TH_V_C  (-0.172f)
#define TH_U_C  (0.529f)
#define TH_R_C  (0.021f)
#define TH_S_C  (0.132f)
#define NEURON_VTH_C 0.5f
#define SPIKE_TS 8

// ---------------- utility kernels ----------------

__global__ void zero_f32_kernel(float* p, long n) {
    long i = (long)blockIdx.x * blockDim.x + threadIdx.x;
    long stride = (long)gridDim.x * blockDim.x;
    for (; i < n; i += stride) p[i] = 0.0f;
}

__global__ void f32_to_bf16_kernel(const float* __restrict__ src, __bf16* __restrict__ dst, int n) {
    int i = blockIdx.x * blockDim.x + threadIdx.x;
    int stride = gridDim.x * blockDim.x;
    for (; i < n; i += stride) dst[i] = (__bf16)src[i];
}

// ---------------- encoder: gaussian receptive field + 8-step spike encode ----------------

__global__ void encoder_kernel(const float* __restrict__ obs, const float* __restrict__ mean,
                               const float* __restrict__ stdv, __bf16* __restrict__ spk,
                               int B, int IN, int OBS, int EP) {
    long i = (long)blockIdx.x * blockDim.x + threadIdx.x;
    long total = (long)B * IN;
    if (i >= total) return;
    int b = (int)(i / IN);
    int f = (int)(i % IN);
    int o = f / EP;
    float d = (obs[(long)b * OBS + o] - mean[f]) / stdv[f];
    float pop = __expf(-0.5f * d * d);
    float volt = 0.0f;
    long base = (long)b * IN + f;
    long step = (long)B * IN;
    #pragma unroll
    for (int t = 0; t < SPIKE_TS; t++) {
        volt += pop;
        float s = (volt > 1.0f) ? 1.0f : 0.0f;
        volt -= s;
        spk[(long)t * step + base] = (__bf16)s;
    }
}

// ---------------- lateral grouped conv on previous output spikes ----------------

__global__ void lateral_kernel(const __bf16* __restrict__ soSpk, const float* __restrict__ conn_w,
                               const float* __restrict__ conn_b, float* __restrict__ lat,
                               int B, int AD, int DP) {
    int OUT = AD * DP;
    int i = blockIdx.x * blockDim.x + threadIdx.x;
    if (i >= B * OUT) return;
    int b = i / OUT, n = i % OUT;
    int a = n / DP, j = n % DP;
    const __bf16* sp = soSpk + (long)b * OUT + a * DP;
    const float*  w  = conn_w + (a * DP + j) * DP;
    float acc = conn_b[n];
    #pragma unroll
    for (int k = 0; k < 10; k++) acc += (float)sp[k] * w[k];
    lat[i] = acc;
}

// ---------------- pop decoder + tanh ----------------

__global__ void decoder_kernel(const float* __restrict__ fr, const float* __restrict__ dec_w,
                               const float* __restrict__ dec_b, float* __restrict__ out,
                               int B, int AD, int DP, float invT) {
    int i = blockIdx.x * blockDim.x + threadIdx.x;
    if (i >= B * AD) return;
    int b = i / AD, a = i % AD;
    const float* f = fr + (long)b * AD * DP + a * DP;
    const float* w = dec_w + a * DP;
    float acc = 0.0f;
    #pragma unroll
    for (int p = 0; p < 10; p++) acc += f[p] * invT * w[p];
    acc += dec_b[a];
    out[i] = tanhf(acc);
}

// ---------------- fused bf16-WMMA GEMM + spiking neuron update ----------------
// C(row b, col n) = A(BxK spikes, bf16) * W^T   (W is (N,K) row-major bf16).
// Wave tile: 32 x (NSUB*16). Block: 256 threads = WAVES_M x WAVES_N waves.
// All tile dims divide the problem exactly (compile-time specialized) -> no guards,
// EXEC stays all-1s, straight-line double-buffered inner loop.

union FragBf16 { v16bf v; uint4 q[2]; };

template<int WAVES_M, int WAVES_N, int NSUB, bool HAS_LAT, bool HAS_FR>
__global__ __launch_bounds__(256)
void gemm_neuron_kernel(const __bf16* __restrict__ A,     // (B x K) input spikes
                        const __bf16* __restrict__ W,     // (N x K) bf16 weights
                        const float*  __restrict__ bias,  // (N)
                        const float*  __restrict__ lat,   // (B x N) extra current
                        float* stC, float* stV, float* stR,
                        __bf16* sSelf,                    // (B x N) prev spikes, overwritten
                        float* fr,                        // (B x N) firing-rate accum
                        int K, int N) {
    const int lane     = threadIdx.x & 31;
    const int wave     = threadIdx.x >> 5;
    const int wm       = wave / WAVES_N;
    const int wn       = wave % WAVES_N;
    const int mBase    = blockIdx.y * (WAVES_M * 32) + wm * 32;
    const int nBase    = blockIdx.x * (WAVES_N * NSUB * 16) + wn * (NSUB * 16);
    const int laneHalf = lane >> 4;       // 0 or 1
    const int lane15   = lane & 15;

    v8f acc[2][NSUB] = {};

    // ISA 7.12.2 lane mapping:
    //  A 16-bit 16x32: lanes0-15 M=lane, K {0..7,16..23}; lanes16-31 M=lane-16, K {8..15,24..31}
    //  B 16-bit 32x16: lanes0-15 N=lane, K {0..7,16..23}; lanes16-31 N=lane-16, K {8..15,24..31}
    const int kA_off = laneHalf ? 8 : 0;
    const int kB_off = laneHalf ? 8 : 0;

    const __bf16* aRow0 = A + (size_t)(mBase + lane15) * K + kA_off;
    const __bf16* aRow1 = aRow0 + (size_t)16 * K;
    const __bf16* bCol[NSUB];
    #pragma unroll
    for (int ns = 0; ns < NSUB; ns++)
        bCol[ns] = W + (size_t)(nBase + ns * 16 + lane15) * K + kB_off;

    FragBf16 af[2][2], bfr[2][NSUB];

    auto loadFrags = [&](auto bufc, int k0) {
        constexpr int buf = decltype(bufc)::value;
        af[buf][0].q[0] = *(const uint4*)(aRow0 + k0);
        af[buf][0].q[1] = *(const uint4*)(aRow0 + k0 + 16);
        af[buf][1].q[0] = *(const uint4*)(aRow1 + k0);
        af[buf][1].q[1] = *(const uint4*)(aRow1 + k0 + 16);
        #pragma unroll
        for (int ns = 0; ns < NSUB; ns++) {
            const uint4* p = (const uint4*)(bCol[ns] + k0);
            bfr[buf][ns].q[0] = p[0];
            bfr[buf][ns].q[1] = p[2];
        }
    };
    auto doMMA = [&](auto bufc) {
        constexpr int buf = decltype(bufc)::value;
        #pragma unroll
        for (int ms = 0; ms < 2; ms++)
            #pragma unroll
            for (int ns = 0; ns < NSUB; ns++)
                acc[ms][ns] = __builtin_amdgcn_wmma_f32_16x16x32_bf16(
                    false, af[buf][ms].v, false, bfr[buf][ns].v,
                    (short)0, acc[ms][ns], false, false);
    };
    using B0 = std::integral_constant<int, 0>;
    using B1 = std::integral_constant<int, 1>;

    // Double-buffered pipeline; requires (K/32) even (holds: 640/32=20, 1024/32=32).
    loadFrags(B0{}, 0);
    int k0 = 0;
    for (; k0 + 64 < K; k0 += 64) {
        loadFrags(B1{}, k0 + 32);
        doMMA(B0{});
        loadFrags(B0{}, k0 + 64);
        doMMA(B1{});
    }
    loadFrags(B1{}, K - 32);
    doMMA(B0{});
    doMMA(B1{});

    // fused neuron epilogue; C/D layout: lanes0-15 VGPR j -> M=j, lanes16-31 -> M=8+j; col = lane15
    #pragma unroll
    for (int ms = 0; ms < 2; ms++) {
        const int rowBase = mBase + ms * 16 + laneHalf * 8;
        #pragma unroll
        for (int ns = 0; ns < NSUB; ns++) {
            const int col = nBase + ns * 16 + lane15;
            const float bb = bias[col];
            #pragma unroll
            for (int j = 0; j < 8; j++) {
                const int row = rowBase + j;
                const size_t idx = (size_t)row * N + col;
                float z = acc[ms][ns][j] + bb;
                if constexpr (HAS_LAT) z += lat[idx];
                float c = stC[idx], v = stV[idx], r = stR[idx];
                float s = (float)sSelf[idx];
                c = c * 0.5f + z;                        // current decay + input
                v = v * (1.0f - s) + s * TH_R_C;         // voltage reset on spike
                r = r + s * TH_S_C;                      // resistance bump
                float vd = v * v - v - r + c;
                float rd = TH_V_C * v - TH_U_C * r;
                v += vd;
                r += rd;
                float sn = (v > NEURON_VTH_C) ? 1.0f : 0.0f;
                stC[idx] = c; stV[idx] = v; stR[idx] = r;
                sSelf[idx] = (__bf16)sn;
                if constexpr (HAS_FR) fr[idx] += sn;
            }
        }
    }
}

// ---------------- host orchestration ----------------

extern "C" void kernel_launch(void* const* d_in, const int* in_sizes, int n_in,
                              void* d_out, int out_size, void* d_ws, size_t ws_size,
                              hipStream_t stream) {
    const float* obs    = (const float*)d_in[0];
    const float* mean   = (const float*)d_in[1];
    const float* stdv   = (const float*)d_in[2];
    const float* w1     = (const float*)d_in[3];
    const float* b1     = (const float*)d_in[4];
    const float* w2     = (const float*)d_in[5];
    const float* b2     = (const float*)d_in[6];
    const float* w_out  = (const float*)d_in[7];
    const float* b_out  = (const float*)d_in[8];
    const float* conn_w = (const float*)d_in[9];
    const float* conn_b = (const float*)d_in[10];
    const float* dec_w  = (const float*)d_in[11];
    const float* dec_b  = (const float*)d_in[12];
    float* out = (float*)d_out;

    const int OBS = 64, EP = 10, AD = 16, DP = 10;
    const int IN = OBS * EP;          // 640
    const int H0 = 1024, H1 = 1024;
    const int OUTN = AD * DP;         // 160
    const int B = in_sizes[0] / OBS;  // 4096

    // ---- workspace layout (256B aligned sections; zero-init prefix first) ----
    char* ws = (char*)d_ws;
    size_t off = 0;
    auto take = [&](size_t bytes) -> char* {
        char* p = ws + off;
        off += (bytes + 255) & ~(size_t)255;
        return p;
    };
    float*  st1  = (float*) take((size_t)3 * B * H0 * 4);   // c,v,r for layer1
    float*  st2  = (float*) take((size_t)3 * B * H1 * 4);
    float*  sto  = (float*) take((size_t)3 * B * OUTN * 4);
    float*  fr   = (float*) take((size_t)B * OUTN * 4);
    __bf16* s1s  = (__bf16*)take((size_t)B * H0 * 2);       // layer spike buffers
    __bf16* s2s  = (__bf16*)take((size_t)B * H1 * 2);
    __bf16* sos  = (__bf16*)take((size_t)B * OUTN * 2);
    const size_t zeroBytes = off;
    __bf16* wb1  = (__bf16*)take((size_t)H0 * IN * 2);
    __bf16* wb2  = (__bf16*)take((size_t)H1 * H0 * 2);
    __bf16* wbo  = (__bf16*)take((size_t)OUTN * H1 * 2);
    __bf16* ins  = (__bf16*)take((size_t)SPIKE_TS * B * IN * 2);
    float*  lat  = (float*) take((size_t)B * OUTN * 4);

    // ---- zero recurrent state + firing-rate accumulators every call ----
    zero_f32_kernel<<<2048, 256, 0, stream>>>((float*)ws, (long)(zeroBytes / 4));

    // ---- fp32 -> bf16 weight conversion ----
    { int n = H0 * IN;   f32_to_bf16_kernel<<<(n + 255) / 256, 256, 0, stream>>>(w1, wb1, n); }
    { int n = H1 * H0;   f32_to_bf16_kernel<<<(n + 255) / 256, 256, 0, stream>>>(w2, wb2, n); }
    { int n = OUTN * H1; f32_to_bf16_kernel<<<(n + 255) / 256, 256, 0, stream>>>(w_out, wbo, n); }

    // ---- encoder: all 8 timesteps of input spikes ----
    {
        long tot = (long)B * IN;
        encoder_kernel<<<(unsigned)((tot + 255) / 256), 256, 0, stream>>>(
            obs, mean, stdv, ins, B, IN, OBS, EP);
    }

    // ---- 8 recurrent timesteps ----
    dim3 blk(256);
    // hidden layers: wave 32x64, block 64x256, grid (N/256, B/64)
    dim3 gh(H0 / 256, B / 64);            // (4, 64)
    // output layer: wave 32x32, block 256x32, grid (160/32, B/256)
    dim3 go(OUTN / 32, B / 256);          // (5, 16)
    float* st1C = st1; float* st1V = st1 + (size_t)B * H0;   float* st1R = st1 + (size_t)2 * B * H0;
    float* st2C = st2; float* st2V = st2 + (size_t)B * H1;   float* st2R = st2 + (size_t)2 * B * H1;
    float* stoC = sto; float* stoV = sto + (size_t)B * OUTN; float* stoR = sto + (size_t)2 * B * OUTN;

    for (int t = 0; t < SPIKE_TS; t++) {
        const __bf16* xin = ins + (size_t)t * B * IN;
        gemm_neuron_kernel<2, 4, 4, false, false><<<gh, blk, 0, stream>>>(
            xin, wb1, b1, nullptr, st1C, st1V, st1R, s1s, nullptr, IN, H0);
        gemm_neuron_kernel<2, 4, 4, false, false><<<gh, blk, 0, stream>>>(
            s1s, wb2, b2, nullptr, st2C, st2V, st2R, s2s, nullptr, H0, H1);
        {
            int tot = B * OUTN;
            lateral_kernel<<<(tot + 255) / 256, 256, 0, stream>>>(sos, conn_w, conn_b, lat, B, AD, DP);
        }
        gemm_neuron_kernel<8, 1, 2, true, true><<<go, blk, 0, stream>>>(
            s2s, wbo, b_out, lat, stoC, stoV, stoR, sos, fr, H1, OUTN);
    }

    // ---- decode ----
    {
        int tot = B * AD;
        decoder_kernel<<<(tot + 255) / 256, 256, 0, stream>>>(
            fr, dec_w, dec_b, out, B, AD, DP, 1.0f / (float)SPIKE_TS);
    }
}
